// PointNet_FeaturePropagation_36120674959458
// MI455X (gfx1250) — matrix operand order repaired and checked by reading, not compiled, not run on hardware
//
#include <hip/hip_runtime.h>

// ---------------- problem constants (from setup_inputs) ----------------
#define B_    4
#define N_    8192
#define S_    2048
#define D1_   128
#define D2_   256
#define CIN_  384
#define O_    256
#define M_    (B_ * N_)   // 32768 rows (b,n flattened)
#define NBLK_ 2

typedef __attribute__((ext_vector_type(16))) __bf16        v16bf;
typedef __attribute__((ext_vector_type(8)))  float         v8f;
typedef __attribute__((ext_vector_type(4)))  unsigned int  u32x4;

__device__ __forceinline__ unsigned short f2bf(float f) {
    union { float f; unsigned u; } v; v.f = f;
    unsigned r = v.u + 0x7FFFu + ((v.u >> 16) & 1u);   // round-to-nearest-even
    return (unsigned short)(r >> 16);
}

// ---------------- 1) transpose points2 (B,D2,S) -> (B,S,D2) fp32 ----------------
__global__ __launch_bounds__(256) void t2_kernel(const float* __restrict__ p2,
                                                 float* __restrict__ p2T) {
    size_t t = (size_t)blockIdx.x * 256 + threadIdx.x;
    if (t >= (size_t)B_ * S_ * D2_) return;
    size_t b = t / ((size_t)S_ * D2_);
    size_t r = t % ((size_t)S_ * D2_);
    size_t s = r / D2_, c = r % D2_;
    p2T[t] = p2[(b * D2_ + c) * S_ + s];
}

// ---------------- 2) 3-NN + inverse-distance weights ----------------
__global__ __launch_bounds__(256) void knn_kernel(const float* __restrict__ xyz1,
                                                  const float* __restrict__ xyz2,
                                                  int* __restrict__ idx,
                                                  float* __restrict__ wgt) {
    __shared__ float sx[S_], sy[S_], sz[S_];
    int b = blockIdx.y;
    int n = blockIdx.x * 256 + threadIdx.x;
    const float* x2 = xyz2 + (size_t)b * 3 * S_;
    for (int s = threadIdx.x; s < S_; s += 256) {
        sx[s] = x2[s]; sy[s] = x2[S_ + s]; sz[s] = x2[2 * S_ + s];
    }
    __syncthreads();
    const float* x1 = xyz1 + (size_t)b * 3 * N_;
    float px = x1[n], py = x1[N_ + n], pz = x1[2 * N_ + n];
    float d0 = 3.4e38f, d1 = 3.4e38f, d2 = 3.4e38f;
    int   i0 = 0, i1 = 0, i2 = 0;
    for (int s = 0; s < S_; ++s) {
        float dx = px - sx[s], dy = py - sy[s], dz = pz - sz[s];
        float d = dx * dx + dy * dy + dz * dz;
        if (d < d2) {
            if (d < d1) {
                if (d < d0) { d2 = d1; i2 = i1; d1 = d0; i1 = i0; d0 = d; i0 = s; }
                else        { d2 = d1; i2 = i1; d1 = d;  i1 = s; }
            } else          { d2 = d;  i2 = s; }
        }
    }
    float r0 = 1.0f / (d0 + 1e-8f), r1 = 1.0f / (d1 + 1e-8f), r2 = 1.0f / (d2 + 1e-8f);
    float inv = 1.0f / (r0 + r1 + r2);
    size_t p = ((size_t)b * N_ + n) * 3;
    idx[p] = i0; idx[p + 1] = i1; idx[p + 2] = i2;
    wgt[p] = r0 * inv; wgt[p + 1] = r1 * inv; wgt[p + 2] = r2 * inv;
}

// ---------------- 3) build concatenated bf16 input (B*N, 384) ----------------
__global__ __launch_bounds__(CIN_) void xcat_kernel(const float* __restrict__ points1,
                                                    const float* __restrict__ p2T,
                                                    const int* __restrict__ idx,
                                                    const float* __restrict__ wgt,
                                                    unsigned short* __restrict__ xcat) {
    size_t pt = blockIdx.x;            // b*N + n
    size_t b = pt / N_, n = pt % N_;
    int c = threadIdx.x;               // 0..383
    float v;
    if (c < D1_) {
        v = points1[(b * D1_ + c) * N_ + n];
    } else {
        int cc = c - D1_;
        const int*   id = idx + pt * 3;
        const float* w  = wgt + pt * 3;
        const float* base = p2T + b * (size_t)S_ * D2_;
        v = w[0] * base[(size_t)id[0] * D2_ + cc]
          + w[1] * base[(size_t)id[1] * D2_ + cc]
          + w[2] * base[(size_t)id[2] * D2_ + cc];
    }
    xcat[pt * CIN_ + c] = f2bf(v);
}

// ---------------- 4) fp32 -> bf16 weight conversion ----------------
__global__ __launch_bounds__(256) void cvt_kernel(const float* __restrict__ src,
                                                  unsigned short* __restrict__ dst, int n) {
    int t = blockIdx.x * 256 + threadIdx.x;
    if (t < n) dst[t] = f2bf(src[t]);
}

// ---------------- 5) bf16 WMMA GEMM + fused BN statistics ----------------
// Block: 8 waves covering 128 M rows x one 64-wide O group.
// Weight panel (64 x K bf16) staged once in LDS; B fragments via ds_load_b128.
// Epilogue: bias add, Y store, per-channel sum/sumsq reduced via shfl + LDS
// ds_add_f32 + 128 global fp32 atomics per block.
#define TILE_O 4
__global__ __launch_bounds__(256) void gemm_kernel(const unsigned short* __restrict__ A,
                                                   const unsigned short* __restrict__ W,
                                                   const float* __restrict__ bias,
                                                   float* __restrict__ Y,
                                                   float* __restrict__ stats,
                                                   int K) {
    __shared__ __align__(16) unsigned short sW[64 * CIN_];   // up to 48 KB
    __shared__ float ssum[64], ssq[64];

    const int numOG = O_ / 64;                     // 4
    const int mb = blockIdx.x / numOG;
    const int o0 = (blockIdx.x % numOG) * 64;
    const int wave = threadIdx.x >> 5;
    const int lane = threadIdx.x & 31;
    const int m0 = (mb * 8 + wave) * 16;
    const int half = lane >> 4;                    // 0 | 1
    const int l16  = lane & 15;

    // cooperative stage of weight panel: rows o0..o0+63 (each K bf16)
    const int chunksPerRow = K / 8;                // u32x4 chunks (8 ushorts)
    for (int c = threadIdx.x; c < 64 * chunksPerRow; c += 256) {
        int row = c / chunksPerRow, col = (c % chunksPerRow) * 8;
        *(u32x4*)(sW + row * K + col) = *(const u32x4*)(W + (size_t)(o0 + row) * K + col);
    }
    if (threadIdx.x < 64) { ssum[threadIdx.x] = 0.f; ssq[threadIdx.x] = 0.f; }
    __syncthreads();

    // A fragment (16-bit A 16x32): lanes 0-15 -> K chunks {0..7, 16..23}
    //                              lanes16-31 -> K chunks {8..15, 24..31}
    const unsigned short* Arow = A + (size_t)(m0 + l16) * K + half * 8;

    v8f acc[TILE_O] = {};
    for (int k = 0; k < K; k += 32) {
        union { v16bf v; u32x4 u[2]; } a;
        a.u[0] = *(const u32x4*)(Arow + k);
        a.u[1] = *(const u32x4*)(Arow + k + 16);
#pragma unroll
        for (int t = 0; t < TILE_O; ++t) {
            // B fragment (16-bit B 32x16): lane -> column o = l16,
            // lanes 0-15 hold K {0..7, 16..23}, lanes 16-31 hold K {8..15, 24..31}
            const unsigned short* Brow = sW + (t * 16 + l16) * K + k + half * 8;
            union { v16bf v; u32x4 u[2]; } bmat;
            bmat.u[0] = *(const u32x4*)(Brow);
            bmat.u[1] = *(const u32x4*)(Brow + 16);
            acc[t] = __builtin_amdgcn_wmma_f32_16x16x32_bf16(
                false, a.v, false, bmat.v, (short)0, acc[t], false, false);
        }
    }

    // epilogue: bias + store + per-channel partial stats
#pragma unroll
    for (int t = 0; t < TILE_O; ++t) {
        int o = o0 + t * 16 + l16;
        float bs = bias[o];
        float s = 0.f, q = 0.f;
#pragma unroll
        for (int i = 0; i < 8; ++i) {   // D VGPR i: M = m0 + i + half*8, N = l16
            size_t m = (size_t)m0 + i + half * 8;
            float v = acc[t][i] + bs;
            Y[m * O_ + o] = v;
            s += v; q += v * v;
        }
        s += __shfl_xor(s, 16, 32);     // combine the two half-wave M groups
        q += __shfl_xor(q, 16, 32);
        if (lane < 16) {
            atomicAdd(&ssum[t * 16 + l16], s);   // ds_add_f32
            atomicAdd(&ssq[t * 16 + l16], q);
        }
    }
    __syncthreads();
    if (threadIdx.x < 64) {
        atomicAdd(&stats[o0 + threadIdx.x],      ssum[threadIdx.x]);
        atomicAdd(&stats[O_ + o0 + threadIdx.x], ssq[threadIdx.x]);
    }
}

__global__ void zstats_kernel(float* __restrict__ stats) { stats[threadIdx.x] = 0.f; }

// ---------------- 6) BN epilogue: scale/shift (+residual) + ReLU, fan-out ----------------
#define ROWS_PER 128
__global__ __launch_bounds__(O_) void bn_kernel(const float* __restrict__ Y,
                                                const float* __restrict__ stats,
                                                const float* __restrict__ g,
                                                const float* __restrict__ be,
                                                const float* __restrict__ resIn,
                                                unsigned short* __restrict__ actOut,
                                                float* __restrict__ resOut,
                                                float* __restrict__ finalOut) {
    int o = threadIdx.x;
    float mean = stats[o] * (1.0f / M_);
    float var  = stats[O_ + o] * (1.0f / M_) - mean * mean;
    float a    = g[o] * rsqrtf(var + 1e-5f);
    float bsh  = be[o] - mean * a;
    size_t r0 = (size_t)blockIdx.x * ROWS_PER;
    for (int r = 0; r < ROWS_PER; ++r) {
        size_t m = r0 + r, off = m * O_ + o;
        float z = Y[off] * a + bsh;
        if (resIn)  z += resIn[off];
        z = fmaxf(z, 0.0f);
        if (actOut)  actOut[off] = f2bf(z);
        if (resOut)  resOut[off] = z;
        if (finalOut) {
            size_t bb = m / N_, n = m % N_;
            finalOut[(bb * O_ + o) * (size_t)N_ + n] = z;   // (B, O, N)
        }
    }
}

// ---------------- workspace layout (bytes) ----------------
#define OFF_P2T   ((size_t)0)                       //  8 MB  fp32 (B,S,D2)
#define OFF_IDX   ((size_t)8388608)                 //  384 KB
#define OFF_WGT   ((size_t)8781824)                 //  384 KB
#define OFF_WFB   ((size_t)9175040)                 //  fuse_W bf16 (196608 B)
#define OFF_W1B   ((size_t)9371648)                 //  blk_W1 bf16 (262144 B)
#define OFF_W2B   ((size_t)9633792)                 //  blk_W2 bf16 (262144 B)
#define OFF_STATS ((size_t)9895936)                 //  2 KB
#define OFF_XCAT  ((size_t)16777216)                // 24 MB bf16 (dead after GEMM0)
#define OFF_ACT   ((size_t)16777216)                // 16 MB bf16 (recycles xcat)
#define OFF_TBF   ((size_t)33554432)                // 16 MB bf16
#define OFF_Y     ((size_t)50331648)                // 32 MB fp32
#define OFF_RES   ((size_t)83886080)                // 32 MB fp32

extern "C" void kernel_launch(void* const* d_in, const int* in_sizes, int n_in,
                              void* d_out, int out_size, void* d_ws, size_t ws_size,
                              hipStream_t stream) {
    const float* xyz1    = (const float*)d_in[0];
    const float* xyz2    = (const float*)d_in[1];
    const float* points1 = (const float*)d_in[2];
    const float* points2 = (const float*)d_in[3];
    const float* fuse_W  = (const float*)d_in[4];
    const float* fuse_b  = (const float*)d_in[5];
    const float* fuse_g  = (const float*)d_in[6];
    const float* fuse_be = (const float*)d_in[7];
    const float* blk_W1  = (const float*)d_in[8];
    const float* blk_b1  = (const float*)d_in[9];
    const float* blk_g1  = (const float*)d_in[10];
    const float* blk_be1 = (const float*)d_in[11];
    const float* blk_W2  = (const float*)d_in[12];
    const float* blk_b2  = (const float*)d_in[13];
    const float* blk_g2  = (const float*)d_in[14];
    const float* blk_be2 = (const float*)d_in[15];

    char* ws = (char*)d_ws;
    float*          p2T   = (float*)(ws + OFF_P2T);
    int*            idx   = (int*)(ws + OFF_IDX);
    float*          wgt   = (float*)(ws + OFF_WGT);
    unsigned short* wfb   = (unsigned short*)(ws + OFF_WFB);
    unsigned short* w1b   = (unsigned short*)(ws + OFF_W1B);
    unsigned short* w2b   = (unsigned short*)(ws + OFF_W2B);
    float*          stats = (float*)(ws + OFF_STATS);
    unsigned short* xcat  = (unsigned short*)(ws + OFF_XCAT);
    unsigned short* act   = (unsigned short*)(ws + OFF_ACT);
    unsigned short* tbf   = (unsigned short*)(ws + OFF_TBF);
    float*          yb    = (float*)(ws + OFF_Y);
    float*          res   = (float*)(ws + OFF_RES);
    float*          out   = (float*)d_out;

    // interpolation front-end
    t2_kernel<<<(B_ * S_ * D2_ + 255) / 256, 256, 0, stream>>>(points2, p2T);
    knn_kernel<<<dim3(N_ / 256, B_), 256, 0, stream>>>(xyz1, xyz2, idx, wgt);
    xcat_kernel<<<M_, CIN_, 0, stream>>>(points1, p2T, idx, wgt, xcat);

    // weight conversion
    cvt_kernel<<<(O_ * CIN_ + 255) / 256, 256, 0, stream>>>(fuse_W, wfb, O_ * CIN_);
    cvt_kernel<<<(NBLK_ * O_ * O_ + 255) / 256, 256, 0, stream>>>(blk_W1, w1b, NBLK_ * O_ * O_);
    cvt_kernel<<<(NBLK_ * O_ * O_ + 255) / 256, 256, 0, stream>>>(blk_W2, w2b, NBLK_ * O_ * O_);

    const int gemmBlocks = (M_ / 128) * (O_ / 64);   // 256 * 4 = 1024
    const int redBlocks  = M_ / ROWS_PER;            // 256

    // ---- fuse stage: x = relu(bn(conv(xcat))) ; x is residual base + gemm input
    zstats_kernel<<<1, 2 * O_, 0, stream>>>(stats);
    gemm_kernel<<<gemmBlocks, 256, 0, stream>>>(xcat, wfb, fuse_b, yb, stats, CIN_);
    bn_kernel<<<redBlocks, O_, 0, stream>>>(yb, stats, fuse_g, fuse_be,
                                            nullptr, act, res, nullptr);

    // ---- residual blocks
    for (int i = 0; i < NBLK_; ++i) {
        // y = relu(bn(conv1(x)))
        zstats_kernel<<<1, 2 * O_, 0, stream>>>(stats);
        gemm_kernel<<<gemmBlocks, 256, 0, stream>>>(act, w1b + (size_t)i * O_ * O_,
                                                    blk_b1 + i * O_, yb, stats, O_);
        bn_kernel<<<redBlocks, O_, 0, stream>>>(yb, stats, blk_g1 + i * O_, blk_be1 + i * O_,
                                                nullptr, tbf, nullptr, nullptr);
        // x = relu(bn(conv2(y)) + x)
        zstats_kernel<<<1, 2 * O_, 0, stream>>>(stats);
        gemm_kernel<<<gemmBlocks, 256, 0, stream>>>(tbf, w2b + (size_t)i * O_ * O_,
                                                    blk_b2 + i * O_, yb, stats, O_);
        if (i < NBLK_ - 1) {
            bn_kernel<<<redBlocks, O_, 0, stream>>>(yb, stats, blk_g2 + i * O_, blk_be2 + i * O_,
                                                    res, act, res, nullptr);
        } else {
            bn_kernel<<<redBlocks, O_, 0, stream>>>(yb, stats, blk_g2 + i * O_, blk_be2 + i * O_,
                                                    res, nullptr, nullptr, out);
        }
    }
}
